// Net_58033598104034
// MI455X (gfx1250) — matrix operand order repaired and checked by reading, compiled, not run on hardware
//
#include <hip/hip_runtime.h>
#include <hip/hip_bf16.h>
#include <math.h>

typedef _Float16 h16;
typedef __attribute__((ext_vector_type(16))) _Float16 v16h;
typedef __attribute__((ext_vector_type(8)))  float    v8f;

#define NN 50000
#define EE 1000000
#define BN_EPS 1e-5f

// ---------- fragment-order LDS indexing (CDNA5 ISA 7.12.2 inverse maps) ----------
// A 16x32 f16 element (m,k) -> lane = ((k>>3)&1)*16 + m ; e = (k>>4)*8 + (k&7)
__device__ __forceinline__ int a_idx(int m, int k) {
  return ((((k >> 3) & 1) << 4) | m) * 16 + (((k >> 4) << 3) | (k & 7));
}
// B 32x16 f16 element (k,n) -> lane = ((k>>3)&1)*16 + n ; e = (k>>4)*8 + (k&7)
__device__ __forceinline__ int b_idx(int k, int n) {
  return ((((k >> 3) & 1) << 4) | n) * 16 + (((k >> 4) << 3) | (k & 7));
}
__device__ __forceinline__ v16h ld_frag(const h16* p) { return *(const v16h*)p; }
__device__ __forceinline__ void st_pair(h16* p, h16 a, h16 b) {
  union { h16 h[2]; unsigned u; } v; v.h[0] = a; v.h[1] = b;
  *(unsigned*)p = v.u;
}
// max for non-negative floats via monotone uint bit pattern
__device__ __forceinline__ void atomicMaxPosF(float* addr, float v) {
  if (v > 0.0f) atomicMax((unsigned int*)addr, __float_as_uint(v));
}

// ---------- utility kernels ----------
__global__ void k_zero(float4* p, int n4) {
  int i = blockIdx.x * blockDim.x + threadIdx.x;
  if (i < n4) p[i] = make_float4(0.f, 0.f, 0.f, 0.f);
}

__global__ void k_bn_finalize(const float* gsum, const float* gsq,
                              const float* gamma, const float* beta,
                              float* scale, float* shift, int C, float invCnt) {
  int c = blockIdx.x * blockDim.x + threadIdx.x;
  if (c >= C) return;
  float mean = gsum[c] * invCnt;
  float var  = fmaxf(gsq[c] * invCnt - mean * mean, 0.f);
  float sc   = gamma[c] * rsqrtf(var + BN_EPS);
  scale[c] = sc;
  shift[c] = beta[c] - mean * sc;
}

// ---------- conv1 layer1 stats: h1 = relu([xi, xj-xi] @ W1 + b1), 6->64 ----------
__global__ void k_c1l1_stats(const float* __restrict__ x, const int* __restrict__ src,
                             const int* __restrict__ dst, const float* __restrict__ W1,
                             const float* __restrict__ b1, float* gsum, float* gsq) {
  int c = threadIdx.x;  // 64 threads = 64 channels
  float w0 = W1[c], w1 = W1[64 + c], w2 = W1[128 + c];
  float w3 = W1[192 + c], w4 = W1[256 + c], w5 = W1[320 + c];
  float bb = b1[c], s = 0.f, q = 0.f;
  for (int e = blockIdx.x; e < EE; e += gridDim.x) {
    int d = dst[e], si = src[e];
    float f0 = x[3 * d], f1 = x[3 * d + 1], f2 = x[3 * d + 2];
    float f3 = x[3 * si] - f0, f4 = x[3 * si + 1] - f1, f5 = x[3 * si + 2] - f2;
    float h = bb + f0 * w0 + f1 * w1 + f2 * w2 + f3 * w3 + f4 * w4 + f5 * w5;
    h = fmaxf(h, 0.f);
    s += h; q += h * h;
  }
  atomicAdd(&gsum[c], s);
  atomicAdd(&gsq[c], q);
}

// shared builder: h1n (BN'd layer-1 output, f16) for 128 edges into frag-order LDS
__device__ __forceinline__ void build_h1n_frag(
    h16 Af[8][2][512], const float* __restrict__ x, const int* __restrict__ src,
    const int* __restrict__ dst, const float* W1s, const float* b1s,
    const float* sc1s, const float* sh1s, int e0, int t) {
  int r = t >> 1, cb = (t & 1) * 32, e = e0 + r;
  h16* f = &Af[r >> 4][cb >> 5][0];
  int m = r & 15;
  if (e < EE) {
    int d = dst[e], si = src[e];
    float f0 = x[3 * d], f1 = x[3 * d + 1], f2 = x[3 * d + 2];
    float f3 = x[3 * si] - f0, f4 = x[3 * si + 1] - f1, f5 = x[3 * si + 2] - f2;
#pragma unroll
    for (int c = 0; c < 32; c += 2) {
      int c0 = cb + c, c1 = cb + c + 1;
      float h0 = b1s[c0] + f0 * W1s[c0] + f1 * W1s[64 + c0] + f2 * W1s[128 + c0]
               + f3 * W1s[192 + c0] + f4 * W1s[256 + c0] + f5 * W1s[320 + c0];
      float h1 = b1s[c1] + f0 * W1s[c1] + f1 * W1s[64 + c1] + f2 * W1s[128 + c1]
               + f3 * W1s[192 + c1] + f4 * W1s[256 + c1] + f5 * W1s[320 + c1];
      st_pair(&f[a_idx(m, c)], (h16)(fmaxf(h0, 0.f) * sc1s[c0] + sh1s[c0]),
                               (h16)(fmaxf(h1, 0.f) * sc1s[c1] + sh1s[c1]));
    }
  } else {
#pragma unroll
    for (int c = 0; c < 32; c += 2) st_pair(&f[a_idx(m, c)], (h16)0.f, (h16)0.f);
  }
}

// load 64x64 f32 weight into frag-order f16 LDS (paired-k stores)
__device__ __forceinline__ void build_w64_frag(h16 Wf[2][4][512],
                                               const float* __restrict__ W, int t) {
  for (int p = t; p < 2048; p += 256) {
    int c = p & 63, k2 = (p >> 6) << 1;
    st_pair(&Wf[k2 >> 5][c >> 4][b_idx(k2 & 31, c & 15)],
            (h16)W[k2 * 64 + c], (h16)W[(k2 + 1) * 64 + c]);
  }
}

// ---------- conv1 layer2 stats: recompute h1n, h2 = relu(h1n @ W2 + b2) ----------
__global__ __launch_bounds__(256) void k_c1l2(
    const float* __restrict__ x, const int* __restrict__ src, const int* __restrict__ dst,
    const float* W1, const float* b1, const float* sc1, const float* sh1,
    const float* W2, const float* b2, float* gsum, float* gsq) {
  __shared__ __align__(32) h16 Af[8][2][512];  // 128 x 64, fragment order
  __shared__ __align__(32) h16 Wf[2][4][512];  // 64 x 64
  __shared__ float W1s[384], b1s[64], sc1s[64], sh1s[64], ssum[64], ssq[64];
  int t = threadIdx.x;
  for (int i = t; i < 384; i += 256) W1s[i] = W1[i];
  if (t < 64) { b1s[t] = b1[t]; sc1s[t] = sc1[t]; sh1s[t] = sh1[t]; ssum[t] = 0.f; ssq[t] = 0.f; }
  build_w64_frag(Wf, W2, t);
  __syncthreads();
  int e0 = blockIdx.x * 128;
  build_h1n_frag(Af, x, src, dst, W1s, b1s, sc1s, sh1s, e0, t);
  __syncthreads();
  int wave = t >> 5, lane = t & 31, hs = lane >> 4, n = lane & 15;
  bool full = (e0 + 128 <= EE);
  for (int ct = 0; ct < 4; ++ct) {
    v8f acc = {};
#pragma unroll
    for (int kb = 0; kb < 2; ++kb) {
      v16h a  = ld_frag(&Af[wave][kb][lane * 16]);
      v16h bf = ld_frag(&Wf[kb][ct][lane * 16]);
      acc = __builtin_amdgcn_wmma_f32_16x16x32_f16(false, a, false, bf, (short)0, acc, false, false);
    }
    int col = ct * 16 + n;
    float bb = b2[col], s = 0.f, q = 0.f;
    if (full) {
#pragma unroll
      for (int r = 0; r < 8; ++r) {
        float h = fmaxf(acc[r] + bb, 0.f);
        s += h; q += h * h;
      }
    } else {
#pragma unroll
      for (int r = 0; r < 8; ++r) {
        int e = e0 + wave * 16 + hs * 8 + r;
        if (e < EE) { float h = fmaxf(acc[r] + bb, 0.f); s += h; q += h * h; }
      }
    }
    atomicAdd(&ssum[col], s);
    atomicAdd(&ssq[col], q);
  }
  __syncthreads();
  if (t < 64) { atomicAdd(&gsum[t], ssum[t]); atomicAdd(&gsq[t], ssq[t]); }
}

// ---------- conv1 layer3: h3 = relu(h2n@W3+b3); stats + seg-max ----------
__global__ __launch_bounds__(256) void k_c1l3(
    const float* __restrict__ x, const int* __restrict__ src, const int* __restrict__ dst,
    const float* W1, const float* b1, const float* sc1, const float* sh1,
    const float* W2, const float* b2, const float* sc2, const float* sh2,
    const float* W3, const float* b3, float* gsum, float* gsq, float* m3) {
  __shared__ __align__(32) h16 Af[8][2][512];
  __shared__ __align__(32) h16 A2f[8][2][512];
  __shared__ __align__(32) h16 Wf[2][4][512];
  __shared__ float W1s[384], b1s[64], sc1s[64], sh1s[64], ssum[64], ssq[64];
  __shared__ int dsts[128];
  int t = threadIdx.x;
  for (int i = t; i < 384; i += 256) W1s[i] = W1[i];
  if (t < 64) { b1s[t] = b1[t]; sc1s[t] = sc1[t]; sh1s[t] = sh1[t]; ssum[t] = 0.f; ssq[t] = 0.f; }
  build_w64_frag(Wf, W2, t);
  int e0 = blockIdx.x * 128;
  if (t < 128) dsts[t] = (e0 + t < EE) ? dst[e0 + t] : 0;
  __syncthreads();
  build_h1n_frag(Af, x, src, dst, W1s, b1s, sc1s, sh1s, e0, t);
  __syncthreads();
  int wave = t >> 5, lane = t & 31, hs = lane >> 4, n = lane & 15;
  // stage 1: h2n into A2f (frag order for next stage)
  for (int ct = 0; ct < 4; ++ct) {
    v8f acc = {};
#pragma unroll
    for (int kb = 0; kb < 2; ++kb) {
      v16h a  = ld_frag(&Af[wave][kb][lane * 16]);
      v16h bf = ld_frag(&Wf[kb][ct][lane * 16]);
      acc = __builtin_amdgcn_wmma_f32_16x16x32_f16(false, a, false, bf, (short)0, acc, false, false);
    }
    int col = ct * 16 + n;
    float bb = b2[col], s2 = sc2[col], h2s = sh2[col];
    h16* f = &A2f[wave][col >> 5][0];
#pragma unroll
    for (int r = 0; r < 8; ++r) {
      int row = hs * 8 + r;  // within 16-row tile
      f[a_idx(row, col & 31)] = (h16)(fmaxf(acc[r] + bb, 0.f) * s2 + h2s);
    }
  }
  __syncthreads();
  build_w64_frag(Wf, W3, t);
  __syncthreads();
  // stage 2: h3, stats, segment max (pre-BN; BN monotone since gamma=1>0)
  bool full = (e0 + 128 <= EE);
  for (int ct = 0; ct < 4; ++ct) {
    v8f acc = {};
#pragma unroll
    for (int kb = 0; kb < 2; ++kb) {
      v16h a  = ld_frag(&A2f[wave][kb][lane * 16]);
      v16h bf = ld_frag(&Wf[kb][ct][lane * 16]);
      acc = __builtin_amdgcn_wmma_f32_16x16x32_f16(false, a, false, bf, (short)0, acc, false, false);
    }
    int col = ct * 16 + n;
    float bb = b3[col], s = 0.f, q = 0.f;
#pragma unroll
    for (int r = 0; r < 8; ++r) {
      int row = wave * 16 + hs * 8 + r;
      if (full || (e0 + row < EE)) {
        float h = fmaxf(acc[r] + bb, 0.f);
        s += h; q += h * h;
        atomicMaxPosF(&m3[dsts[row] * 64 + col], h);
      }
    }
    atomicAdd(&ssum[col], s);
    atomicAdd(&ssq[col], q);
  }
  __syncthreads();
  if (t < 64) { atomicAdd(&gsum[t], ssum[t]); atomicAdd(&gsq[t], ssq[t]); }
}

// ---------- apply BN3 to pooled m3 -> x1 (f16 copies for conv2 + lin1) ----------
__global__ void k_x1(const float* __restrict__ m3, const float* sc, const float* sh,
                     h16* x1h, h16* feat_h) {
  int i = blockIdx.x * blockDim.x + threadIdx.x;
  if (i >= NN * 64) return;
  int nd = i >> 6, c = i & 63;
  float v = m3[i] * sc[c] + sh[c];
  x1h[i] = (h16)v;
  feat_h[nd * 192 + c] = (h16)v;
}

// ---------- conv2: feat=[x1_i, x1_j - x1_i] (128) -> relu(@W+b); stats + seg-max ----------
__global__ __launch_bounds__(256) void k_c2(
    const h16* __restrict__ x1h, const int* __restrict__ src, const int* __restrict__ dst,
    const float* __restrict__ W, const float* b, float* gsum, float* gsq, float* mc2) {
  __shared__ __align__(32) h16 Af[4][4][512];  // 64 x 128
  __shared__ __align__(32) h16 Wf[4][8][512];  // 128 x 128
  __shared__ float ssum[128], ssq[128];
  __shared__ int dsts[64];
  int t = threadIdx.x;
  for (int p = t; p < 8192; p += 256) {
    int c = p & 127, k2 = (p >> 7) << 1;
    st_pair(&Wf[k2 >> 5][c >> 4][b_idx(k2 & 31, c & 15)],
            (h16)W[k2 * 128 + c], (h16)W[(k2 + 1) * 128 + c]);
  }
  if (t < 128) { ssum[t] = 0.f; ssq[t] = 0.f; }
  int e0 = blockIdx.x * 64;  // EE % 64 == 0: all blocks full
  if (t < 64) dsts[t] = dst[e0 + t];
  {
    int r = t >> 2, cb = (t & 3) * 32, e = e0 + r;
    int d = dst[e], si = src[e];
    h16* f = &Af[r >> 4][cb >> 5][0];
    int m = r & 15;
#pragma unroll
    for (int c = 0; c < 32; c += 2) {
      int cc = cb + c;
      h16 v0, v1;
      if (cc < 64) {
        v0 = x1h[d * 64 + cc]; v1 = x1h[d * 64 + cc + 1];
      } else {
        v0 = (h16)((float)x1h[si * 64 + cc - 64] - (float)x1h[d * 64 + cc - 64]);
        v1 = (h16)((float)x1h[si * 64 + cc - 63] - (float)x1h[d * 64 + cc - 63]);
      }
      st_pair(&f[a_idx(m, c)], v0, v1);
    }
  }
  __syncthreads();
  int wave = t >> 5, lane = t & 31, hs = lane >> 4, n = lane & 15;
  for (int tt = 0; tt < 4; ++tt) {
    int tid = wave * 4 + tt, rt = tid >> 3, ct = tid & 7;
    v8f acc = {};
#pragma unroll
    for (int kb = 0; kb < 4; ++kb) {
      v16h a  = ld_frag(&Af[rt][kb][lane * 16]);
      v16h bf = ld_frag(&Wf[kb][ct][lane * 16]);
      acc = __builtin_amdgcn_wmma_f32_16x16x32_f16(false, a, false, bf, (short)0, acc, false, false);
    }
    int col = ct * 16 + n;
    float bb = b[col], s = 0.f, q = 0.f;
#pragma unroll
    for (int r = 0; r < 8; ++r) {
      int row = rt * 16 + hs * 8 + r;
      float h = fmaxf(acc[r] + bb, 0.f);
      s += h; q += h * h;
      atomicMaxPosF(&mc2[dsts[row] * 128 + col], h);
    }
    atomicAdd(&ssum[col], s);
    atomicAdd(&ssq[col], q);
  }
  __syncthreads();
  if (t < 128) { atomicAdd(&gsum[t], ssum[t]); atomicAdd(&gsq[t], ssq[t]); }
}

// ---------- apply BN_c2 to pooled mc2 -> x2 part of feat ----------
__global__ void k_x2(const float* __restrict__ mc2, const float* sc, const float* sh,
                     h16* feat_h) {
  int i = blockIdx.x * blockDim.x + threadIdx.x;
  if (i >= NN * 128) return;
  int nd = i >> 7, c = i & 127;
  feat_h[nd * 192 + 64 + c] = (h16)(mc2[i] * sc[c] + sh[c]);
}

// ---------- lin1: [N,192] @ [192,1024], relu, stats + batch max-pool ----------
__global__ __launch_bounds__(256) void k_lin1(
    const h16* __restrict__ feat_h, const int* __restrict__ batch,
    const float* __restrict__ W, const float* b, float* gsum, float* gsq, float* pooled) {
  __shared__ __align__(32) h16 Af[6][512];     // 16 x 192
  __shared__ __align__(32) h16 Wf[6][8][512];  // 192 x 128
  __shared__ float ssum[128], ssq[128];
  __shared__ int bs[16];
  int t = threadIdx.x;
  int n0 = blockIdx.x * 16, cb0 = blockIdx.y * 128;  // NN % 16 == 0: all full
  if (t < 16) bs[t] = batch[n0 + t];
  for (int p = t; p < 1536; p += 256) {  // A: paired contiguous loads+stores
    int r = p / 96, c2 = (p % 96) * 2;
    h16 v0 = feat_h[(n0 + r) * 192 + c2], v1 = feat_h[(n0 + r) * 192 + c2 + 1];
    st_pair(&Af[c2 >> 5][a_idx(r, c2 & 31)], v0, v1);
  }
  for (int p = t; p < 12288; p += 256) {
    int c = p & 127, k2 = (p >> 7) << 1;
    st_pair(&Wf[k2 >> 5][c >> 4][b_idx(k2 & 31, c & 15)],
            (h16)W[k2 * 1024 + cb0 + c], (h16)W[(k2 + 1) * 1024 + cb0 + c]);
  }
  if (t < 128) { ssum[t] = 0.f; ssq[t] = 0.f; }
  __syncthreads();
  int wave = t >> 5, lane = t & 31, hs = lane >> 4, n = lane & 15;
  int ct = wave;  // 8 waves cover 8 column subtiles
  v8f acc = {};
#pragma unroll
  for (int kb = 0; kb < 6; ++kb) {
    v16h a  = ld_frag(&Af[kb][lane * 16]);
    v16h bf = ld_frag(&Wf[kb][ct][lane * 16]);
    acc = __builtin_amdgcn_wmma_f32_16x16x32_f16(false, a, false, bf, (short)0, acc, false, false);
  }
  int col = cb0 + ct * 16 + n;
  float bb = b[col], s = 0.f, q = 0.f;
#pragma unroll
  for (int r = 0; r < 8; ++r) {
    int nd = hs * 8 + r;
    float h = fmaxf(acc[r] + bb, 0.f);
    s += h; q += h * h;
    atomicMaxPosF(&pooled[bs[nd] * 1024 + col], h);
  }
  atomicAdd(&ssum[ct * 16 + n], s);
  atomicAdd(&ssq[ct * 16 + n], q);
  __syncthreads();
  if (t < 128) { atomicAdd(&gsum[cb0 + t], ssum[t]); atomicAdd(&gsq[cb0 + t], ssq[t]); }
}

// ---------- small head kernels (32 rows, VALU f32) ----------
__global__ void k_pooled_bn(const float* pooled, const float* sc, const float* sh,
                            float* out) {
  int i = blockIdx.x * blockDim.x + threadIdx.x;
  if (i >= 32 * 1024) return;
  int c = i & 1023;
  out[i] = pooled[i] * sc[c] + sh[c];
}

__global__ void k_mlp1(const float* __restrict__ in, const float* __restrict__ W,
                       const float* b, float* out, float* gsum, float* gsq) {
  int o = blockIdx.x * blockDim.x + threadIdx.x;
  if (o >= 32 * 512) return;
  int r = o >> 9, c = o & 511;
  float acc = b[c];
  for (int k = 0; k < 1024; ++k) acc += in[r * 1024 + k] * W[k * 512 + c];
  float h = fmaxf(acc, 0.f);
  out[o] = h;
  atomicAdd(&gsum[c], h);
  atomicAdd(&gsq[c], h * h);
}

__global__ void k_mlp2(const float* __restrict__ in, const float* scIn, const float* shIn,
                       const float* __restrict__ W, const float* b, float* out,
                       float* gsum, float* gsq) {
  int o = blockIdx.x * blockDim.x + threadIdx.x;
  if (o >= 32 * 256) return;
  int r = o >> 8, c = o & 255;
  float acc = b[c];
  for (int k = 0; k < 512; ++k) acc += (in[r * 512 + k] * scIn[k] + shIn[k]) * W[k * 256 + c];
  float h = fmaxf(acc, 0.f);
  out[o] = h;
  atomicAdd(&gsum[c], h);
  atomicAdd(&gsq[c], h * h);
}

__global__ __launch_bounds__(64) void k_final(const float* __restrict__ hm2,
                                              const float* sc, const float* sh,
                                              const float* __restrict__ W, const float* b,
                                              float* out) {
  __shared__ float logits[40];
  __shared__ float red[2];
  int r = blockIdx.x, t = threadIdx.x;
  if (t < 40) {
    float acc = b[t];
    for (int k = 0; k < 256; ++k) acc += (hm2[r * 256 + k] * sc[k] + sh[k]) * W[k * 40 + t];
    logits[t] = acc;
  }
  __syncthreads();
  if (t == 0) {
    float mm = logits[0];
    for (int i = 1; i < 40; ++i) mm = fmaxf(mm, logits[i]);
    float s = 0.f;
    for (int i = 0; i < 40; ++i) s += expf(logits[i] - mm);
    red[0] = mm; red[1] = logf(s);
  }
  __syncthreads();
  if (t < 40) out[r * 40 + t] = logits[t] - red[0] - red[1];
}

extern "C" void kernel_launch(void* const* d_in, const int* in_sizes, int n_in,
                              void* d_out, int out_size, void* d_ws, size_t ws_size,
                              hipStream_t stream) {
  (void)in_sizes; (void)n_in; (void)out_size; (void)ws_size;
  const float* x    = (const float*)d_in[0];
  const int*   ei   = (const int*)d_in[1];
  const int*   src  = ei;
  const int*   dstp = ei + EE;
  const int*   batch = (const int*)d_in[2];
  int idx = 3;
  const float *c1W[3], *c1b[3], *c1g[3], *c1be[3];
  for (int l = 0; l < 3; ++l) {
    c1W[l]  = (const float*)d_in[idx++]; c1b[l]  = (const float*)d_in[idx++];
    c1g[l]  = (const float*)d_in[idx++]; c1be[l] = (const float*)d_in[idx++];
  }
  const float* c2W  = (const float*)d_in[idx++]; const float* c2b  = (const float*)d_in[idx++];
  const float* c2g  = (const float*)d_in[idx++]; const float* c2be = (const float*)d_in[idx++];
  const float* l1W  = (const float*)d_in[idx++]; const float* l1b  = (const float*)d_in[idx++];
  const float* l1g  = (const float*)d_in[idx++]; const float* l1be = (const float*)d_in[idx++];
  const float* m1W  = (const float*)d_in[idx++]; const float* m1b  = (const float*)d_in[idx++];
  const float* m1g  = (const float*)d_in[idx++]; const float* m1be = (const float*)d_in[idx++];
  const float* m2W  = (const float*)d_in[idx++]; const float* m2b  = (const float*)d_in[idx++];
  const float* m2g  = (const float*)d_in[idx++]; const float* m2be = (const float*)d_in[idx++];
  const float* fW   = (const float*)d_in[idx++]; const float* fb   = (const float*)d_in[idx++];
  float* out = (float*)d_out;

  // ---- workspace layout (zero-required regions first) ----
  float* w = (float*)d_ws;
  size_t o = 0;
  auto alloc = [&](size_t nf) { float* p = w + o; o += nf; return p; };
  float* gsum1 = alloc(64);  float* gsq1 = alloc(64);
  float* gsum2 = alloc(64);  float* gsq2 = alloc(64);
  float* gsum3 = alloc(64);  float* gsq3 = alloc(64);
  float* gsumC = alloc(128); float* gsqC = alloc(128);
  float* gsumL = alloc(1024); float* gsqL = alloc(1024);
  float* gsumM1 = alloc(512); float* gsqM1 = alloc(512);
  float* gsumM2 = alloc(256); float* gsqM2 = alloc(256);
  float* pooled = alloc(32 * 1024);
  float* m3  = alloc((size_t)NN * 64);
  float* mc2 = alloc((size_t)NN * 128);
  size_t zero_floats = o;  // everything above must start at 0 each call
  float* sc1 = alloc(64);  float* sh1 = alloc(64);
  float* sc2 = alloc(64);  float* sh2 = alloc(64);
  float* sc3 = alloc(64);  float* sh3 = alloc(64);
  float* scC = alloc(128); float* shC = alloc(128);
  float* scL = alloc(1024); float* shL = alloc(1024);
  float* scM1 = alloc(512); float* shM1 = alloc(512);
  float* scM2 = alloc(256); float* shM2 = alloc(256);
  float* pooled_bn = alloc(32 * 1024);
  float* hm1 = alloc(32 * 512);
  float* hm2 = alloc(32 * 256);
  h16* x1h    = (h16*)alloc((size_t)NN * 32);  // N*64 halfs
  h16* feat_h = (h16*)alloc((size_t)NN * 96);  // N*192 halfs

  int n4 = (int)(zero_floats / 4);
  k_zero<<<(n4 + 255) / 256, 256, 0, stream>>>((float4*)w, n4);

  // conv1
  k_c1l1_stats<<<2048, 64, 0, stream>>>(x, src, dstp, c1W[0], c1b[0], gsum1, gsq1);
  k_bn_finalize<<<1, 64, 0, stream>>>(gsum1, gsq1, c1g[0], c1be[0], sc1, sh1, 64, 1.0f / EE);
  k_c1l2<<<(EE + 127) / 128, 256, 0, stream>>>(x, src, dstp, c1W[0], c1b[0], sc1, sh1,
                                               c1W[1], c1b[1], gsum2, gsq2);
  k_bn_finalize<<<1, 64, 0, stream>>>(gsum2, gsq2, c1g[1], c1be[1], sc2, sh2, 64, 1.0f / EE);
  k_c1l3<<<(EE + 127) / 128, 256, 0, stream>>>(x, src, dstp, c1W[0], c1b[0], sc1, sh1,
                                               c1W[1], c1b[1], sc2, sh2, c1W[2], c1b[2],
                                               gsum3, gsq3, m3);
  k_bn_finalize<<<1, 64, 0, stream>>>(gsum3, gsq3, c1g[2], c1be[2], sc3, sh3, 64, 1.0f / EE);
  k_x1<<<(NN * 64 + 255) / 256, 256, 0, stream>>>(m3, sc3, sh3, x1h, feat_h);

  // conv2
  k_c2<<<EE / 64, 256, 0, stream>>>(x1h, src, dstp, c2W, c2b, gsumC, gsqC, mc2);
  k_bn_finalize<<<2, 64, 0, stream>>>(gsumC, gsqC, c2g, c2be, scC, shC, 128, 1.0f / EE);
  k_x2<<<(NN * 128 + 255) / 256, 256, 0, stream>>>(mc2, scC, shC, feat_h);

  // lin1 + global max pool
  dim3 gLin(NN / 16, 8);
  k_lin1<<<gLin, 256, 0, stream>>>(feat_h, batch, l1W, l1b, gsumL, gsqL, pooled);
  k_bn_finalize<<<16, 64, 0, stream>>>(gsumL, gsqL, l1g, l1be, scL, shL, 1024, 1.0f / NN);
  k_pooled_bn<<<128, 256, 0, stream>>>(pooled, scL, shL, pooled_bn);

  // head
  k_mlp1<<<64, 256, 0, stream>>>(pooled_bn, m1W, m1b, hm1, gsumM1, gsqM1);
  k_bn_finalize<<<8, 64, 0, stream>>>(gsumM1, gsqM1, m1g, m1be, scM1, shM1, 512, 1.0f / 32);
  k_mlp2<<<32, 256, 0, stream>>>(hm1, scM1, shM1, m2W, m2b, hm2, gsumM2, gsqM2);
  k_bn_finalize<<<4, 64, 0, stream>>>(gsumM2, gsqM2, m2g, m2be, scM2, shM2, 256, 1.0f / 32);
  k_final<<<32, 64, 0, stream>>>(hm2, scM2, shM2, fW, fb, out);
}
